// AttentionBlock_68075231641927
// MI455X (gfx1250) — matrix operand order repaired and checked by reading, not compiled, not run on hardware
//
#include <hip/hip_runtime.h>

// ---------------------------------------------------------------------------
// AttentionBlock for MI455X (gfx1250, wave32, WMMA bf16 16x16x32)
//   N=16, C=512, H=W=32 -> S=1024, GROUPS=32
// v3: fix async global->LDS builtin pointer types (v4i addrspace pointers).
// ---------------------------------------------------------------------------

#define N_   16
#define C_   512
#define S_   1024
#define G_   32
#define CPG_ 16
#define NS_  (N_ * S_)

typedef __bf16 bf16;
typedef __attribute__((ext_vector_type(16))) __bf16 v16bf;
typedef __attribute__((ext_vector_type(8)))  __bf16 v8bf;
typedef __attribute__((ext_vector_type(8)))  float  v8f;
typedef __attribute__((ext_vector_type(4)))  int    v4i;

// --- CDNA5 async global->LDS copy (guarded; falls back to sync copy) -------
#if defined(__has_builtin)
#if __has_builtin(__builtin_amdgcn_global_load_async_to_lds_b128)
#define HAVE_ASYNC_B128 1
#endif
#if __has_builtin(__builtin_amdgcn_s_wait_asynccnt)
#define HAVE_WAIT_ASYNC 1
#endif
#endif

#if defined(HAVE_ASYNC_B128)
typedef __attribute__((address_space(1))) v4i gas_v4i;
typedef __attribute__((address_space(3))) v4i las_v4i;
#endif

static __device__ __forceinline__ void async_copy_b128(const void* g, void* l) {
#if defined(HAVE_ASYNC_B128)
  __builtin_amdgcn_global_load_async_to_lds_b128(
      (gas_v4i*)g, (las_v4i*)l, 0, 0);
#else
  *(v8bf*)l = *(const v8bf*)g;
#endif
}

static __device__ __forceinline__ void async_copy_wait() {
#if defined(HAVE_ASYNC_B128)
#if defined(HAVE_WAIT_ASYNC)
  __builtin_amdgcn_s_wait_asynccnt(0);
#else
  asm volatile("s_wait_asynccnt 0x0" ::: "memory");
#endif
#endif
}

// --- WMMA fragment loaders (contiguous k order, same for A and B, wave32) --------------
// A (16x32 bf16, MxK): lane m=l%16; lanes 0-15 hold K 0-15,
// lanes 16-31 hold K 16-31 -> two 16B loads per lane.
static __device__ __forceinline__ v16bf load_a_frag(const bf16* tile, int ld, int lane) {
  const bf16* p = tile + (size_t)(lane & 15) * ld + ((lane >> 4) << 4);
  v8bf lo = *(const v8bf*)p;
  v8bf hi = *(const v8bf*)(p + 8);
  return __builtin_shufflevector(lo, hi, 0, 1, 2, 3, 4, 5, 6, 7,
                                         8, 9, 10, 11, 12, 13, 14, 15);
}

// B (32x16 bf16, KxN) for A*B^T form: operand stored row-major over K.
// lane n=l%16; lanes 0-15 hold K 0-15, lanes 16-31 hold K 16-31 ->
// one contiguous 32B load per lane.
static __device__ __forceinline__ v16bf load_b_frag(const bf16* tile, int ld, int lane) {
  const bf16* p = tile + (size_t)(lane & 15) * ld + ((lane >> 4) << 4);
  return *(const v16bf*)p;
}

static __device__ __forceinline__ v8f wmma_bf16(v16bf a, v16bf b, v8f c) {
  // v_wmma_f32_16x16x32_bf16  D = A x B + C
  return __builtin_amdgcn_wmma_f32_16x16x32_bf16(
      /*neg_a=*/false, a, /*neg_b=*/false, b,
      /*c_mod=*/(short)0, c, /*reuse_a=*/false, /*reuse_b=*/false);
}

// --- Kernel 0: fp32 -> bf16 weight conversion ------------------------------
__global__ void cvt_w_k(const float* __restrict__ wq, const float* __restrict__ wk,
                        const float* __restrict__ wv, const float* __restrict__ wo,
                        bf16* __restrict__ wqB, bf16* __restrict__ wkB,
                        bf16* __restrict__ wvB, bf16* __restrict__ woB) {
  int i = blockIdx.x * 256 + threadIdx.x;
  if (i < C_ * C_) {
    wqB[i] = (bf16)wq[i];
    wkB[i] = (bf16)wk[i];
    wvB[i] = (bf16)wv[i];
    woB[i] = (bf16)wo[i];
  }
}

// --- Kernel 1: GroupNorm, output transposed to h[N][S][C] in bf16 ----------
__global__ void gn_k(const float* __restrict__ x, const float* __restrict__ gamma,
                     const float* __restrict__ beta, bf16* __restrict__ h) {
  __shared__ float rs[256];
  __shared__ float rq[256];
  int n = blockIdx.x, g = blockIdx.y, tid = threadIdx.x;
  const float* xp = x + ((size_t)n * C_ + (size_t)g * CPG_) * S_;
  float s = 0.f, ss = 0.f;
  for (int i = tid; i < CPG_ * S_; i += 256) {
    float v = xp[i];
    s += v;
    ss += v * v;
  }
  rs[tid] = s;
  rq[tid] = ss;
  __syncthreads();
  for (int off = 128; off > 0; off >>= 1) {
    if (tid < off) { rs[tid] += rs[tid + off]; rq[tid] += rq[tid + off]; }
    __syncthreads();
  }
  const float inv_n = 1.0f / (float)(CPG_ * S_);
  float mean = rs[0] * inv_n;
  float var  = rq[0] * inv_n - mean * mean;
  float rstd = rsqrtf(var + 1e-5f);
  for (int i = tid; i < CPG_ * S_; i += 256) {
    int cc = i >> 10;
    int sp = i & (S_ - 1);
    int c  = g * CPG_ + cc;
    float v = (xp[i] - mean) * rstd * gamma[c] + beta[c];
    h[((size_t)n * S_ + sp) * C_ + c] = (bf16)v;
  }
}

// --- Kernel 2: fused QKV projection GEMMs ----------------------------------
// Weight tiles (shared by all 4 waves) staged once into LDS via async copy.
__global__ void __launch_bounds__(128)
qkv_k(const bf16* __restrict__ h,
      const bf16* __restrict__ wq, const bf16* __restrict__ wk, const bf16* __restrict__ wv,
      const float* __restrict__ bq, const float* __restrict__ bk, const float* __restrict__ bv,
      bf16* __restrict__ q, bf16* __restrict__ k, bf16* __restrict__ vT) {
  __shared__ __attribute__((aligned(32))) bf16 sw[3][16][C_];   // 48 KB
  int tid = threadIdx.x, lane = tid & 31, wave = tid >> 5;
  int s0 = blockIdx.x * 64 + wave * 16;
  int o0 = blockIdx.y * 16;

  // Stage the 16x512 weight tiles of all three projections into LDS.
  for (int ch = tid; ch < 16 * 64; ch += 128) {
    int row = ch >> 6;
    int col = (ch & 63) << 3;
    size_t goff = (size_t)(o0 + row) * C_ + col;
    async_copy_b128(wq + goff, &sw[0][row][col]);
    async_copy_b128(wk + goff, &sw[1][row][col]);
    async_copy_b128(wv + goff, &sw[2][row][col]);
  }
  async_copy_wait();
  __syncthreads();

  v8f aq = {}, ak = {}, av = {};
  const bf16* arow = h + (size_t)s0 * C_;
  v16bf a_cur = load_a_frag(arow, C_, lane);
  for (int c0 = 0; c0 < C_; c0 += 32) {
    v16bf a_nxt = a_cur;
    if (c0 + 32 < C_) {
      __builtin_prefetch(arow + (size_t)(lane & 15) * C_ + c0 + 64, 0, 3);
      a_nxt = load_a_frag(arow + c0 + 32, C_, lane);
    }
    v16bf bQ = load_b_frag(&sw[0][0][0] + c0, C_, lane);   // ds_load
    v16bf bK = load_b_frag(&sw[1][0][0] + c0, C_, lane);
    v16bf bV = load_b_frag(&sw[2][0][0] + c0, C_, lane);
    aq = wmma_bf16(a_cur, bQ, aq);
    ak = wmma_bf16(a_cur, bK, ak);
    av = wmma_bf16(a_cur, bV, av);
    a_cur = a_nxt;
  }
  int ncol = lane & 15;
  int o = o0 + ncol;
  float fq = bq[o], fk = bk[o], fv = bv[o];
  for (int r = 0; r < 8; r++) {
    int m  = r + ((lane >> 4) << 3);   // C/D layout: vgpr r -> row r / r+8
    int sg = s0 + m;
    q[(size_t)sg * C_ + o] = (bf16)(aq[r] + fq);
    k[(size_t)sg * C_ + o] = (bf16)(ak[r] + fk);
    int n = sg >> 10, sl = sg & (S_ - 1);
    vT[((size_t)n * C_ + o) * S_ + sl] = (bf16)(av[r] + fv);
  }
}

// --- Kernel 3: fused attention for one (batch, 16-query-row) strip ---------
// Phase 1: logits(16x1024) = QK^T * scale -> LDS fp32 (Q fragments hoisted
//          into registers once; 2 key tiles in flight per wave)
// Phase 2: row softmax in LDS -> bf16 probs in LDS
// Phase 3: O(16x512) = P*V, one LDS A-fragment feeds 8 accumulator chains
__global__ void __launch_bounds__(128)
attn_k(const bf16* __restrict__ q, const bf16* __restrict__ k,
       const bf16* __restrict__ vT, bf16* __restrict__ ao) {
  __shared__ __attribute__((aligned(32))) float lg[16 * 1024];   // 64 KB logits
  __shared__ __attribute__((aligned(32))) bf16  pr[16 * 1024];   // 32 KB probs
  __shared__ float red[16][9];
  __shared__ float stat[16];
  int tid = threadIdx.x, lane = tid & 31, wave = tid >> 5;
  int n = blockIdx.x;
  int s0 = blockIdx.y * 16;
  const bf16* qrow  = q + ((size_t)n * S_ + s0) * C_;
  const bf16* kbase = k + (size_t)n * S_ * C_;
  const float scale = 0.044194173824159216f;  // 512^-0.5

  // Hoist all 16 Q A-fragments (16x512 strip) into registers.
  v16bf afrag[16];
#pragma unroll
  for (int kc = 0; kc < 16; kc++)
    afrag[kc] = load_a_frag(qrow + kc * 32, C_, lane);

  // Phase 1: two key tiles per iteration -> two independent WMMA chains.
  for (int t4 = wave; t4 < 64; t4 += 8) {
    v8f acc0 = {}, acc1 = {};
    const bf16* k0 = kbase + (size_t)(t4 * 16) * C_;
    const bf16* k1 = kbase + (size_t)((t4 + 4) * 16) * C_;
#pragma unroll
    for (int kc = 0; kc < 16; kc++) {
      v16bf b0 = load_b_frag(k0 + kc * 32, C_, lane);
      v16bf b1 = load_b_frag(k1 + kc * 32, C_, lane);
      acc0 = wmma_bf16(afrag[kc], b0, acc0);
      acc1 = wmma_bf16(afrag[kc], b1, acc1);
    }
    int ncol = lane & 15;
#pragma unroll
    for (int r = 0; r < 8; r++) {
      int m = r + ((lane >> 4) << 3);
      lg[m * 1024 + t4 * 16 + ncol]       = acc0[r] * scale;
      lg[m * 1024 + (t4 + 4) * 16 + ncol] = acc1[r] * scale;
    }
  }
  __syncthreads();

  // Phase 2: softmax, 8 threads per row.
  int row = tid >> 3, sub = tid & 7;
  float* lrow = lg + row * 1024;
  float mx = -3.0e38f;
  for (int j = 0; j < 128; j++) mx = fmaxf(mx, lrow[sub * 128 + j]);
  red[row][sub] = mx;
  __syncthreads();
  if (sub == 0) {
    float m2 = red[row][0];
    for (int j = 1; j < 8; j++) m2 = fmaxf(m2, red[row][j]);
    stat[row] = m2;
  }
  __syncthreads();
  float rmax = stat[row];
  float sum = 0.f;
  for (int j = 0; j < 128; j++) {
    int idx = sub * 128 + j;
    float e = __expf(lrow[idx] - rmax);
    lrow[idx] = e;               // each slot owned by exactly one thread
    sum += e;
  }
  red[row][sub] = sum;
  __syncthreads();
  if (sub == 0) {
    float s2 = 0.f;
    for (int j = 0; j < 8; j++) s2 += red[row][j];
    stat[row] = 1.0f / s2;
  }
  __syncthreads();
  float inv = stat[row];
  for (int j = 0; j < 128; j++) {
    int idx = sub * 128 + j;
    pr[row * 1024 + idx] = (bf16)(lrow[idx] * inv);
  }
  __syncthreads();

  // Phase 3: O = P * V.  vT[n][c][s] makes this A*B^T over t.
  // One LDS A-fragment per k-step feeds 8 independent accumulators.
  const bf16* vbase = vT + (size_t)n * C_ * S_;
  int c0base = wave * 128;          // 8 contiguous 16-wide channel tiles
  v8f acc[8];
#pragma unroll
  for (int j = 0; j < 8; j++) acc[j] = (v8f){};
  for (int t0 = 0; t0 < S_; t0 += 32) {
    v16bf a = load_a_frag(pr + t0, 1024, lane);              // ds_load
#pragma unroll
    for (int j = 0; j < 8; j++) {
      int c0 = c0base + j * 16;
      v16bf b = load_b_frag(vbase + (size_t)c0 * S_ + t0, S_, lane);
      acc[j] = wmma_bf16(a, b, acc[j]);
    }
  }
  int ncol = lane & 15;
#pragma unroll
  for (int j = 0; j < 8; j++) {
    int c0 = c0base + j * 16;
#pragma unroll
    for (int r = 0; r < 8; r++) {
      int m = r + ((lane >> 4) << 3);
      ao[((size_t)n * S_ + s0 + m) * C_ + c0 + ncol] = (bf16)acc[j][r];
    }
  }
}

// --- Kernel 4: output projection + bias + residual, fp32 out [N][C][S] -----
__global__ void __launch_bounds__(128)
proj_k(const bf16* __restrict__ ao, const bf16* __restrict__ wo,
       const float* __restrict__ bo, const float* __restrict__ x,
       float* __restrict__ out) {
  __shared__ __attribute__((aligned(32))) bf16 sw[16][C_];   // 16 KB
  int tid = threadIdx.x, lane = tid & 31, wave = tid >> 5;
  int s0 = blockIdx.x * 64 + wave * 16;
  int o0 = blockIdx.y * 16;

  for (int ch = tid; ch < 16 * 64; ch += 128) {
    int row = ch >> 6;
    int col = (ch & 63) << 3;
    async_copy_b128(wo + (size_t)(o0 + row) * C_ + col, &sw[row][col]);
  }
  async_copy_wait();
  __syncthreads();

  v8f acc = {};
  const bf16* arow = ao + (size_t)s0 * C_;
  v16bf a_cur = load_a_frag(arow, C_, lane);
  for (int c0 = 0; c0 < C_; c0 += 32) {
    v16bf a_nxt = a_cur;
    if (c0 + 32 < C_) {
      __builtin_prefetch(arow + (size_t)(lane & 15) * C_ + c0 + 64, 0, 3);
      a_nxt = load_a_frag(arow + c0 + 32, C_, lane);
    }
    v16bf b = load_b_frag(&sw[0][0] + c0, C_, lane);        // ds_load
    acc = wmma_bf16(a_cur, b, acc);
    a_cur = a_nxt;
  }
  int ncol = lane & 15, o = o0 + ncol;
  float fb = bo[o];
  for (int r = 0; r < 8; r++) {
    int m  = r + ((lane >> 4) << 3);
    int sg = s0 + m;
    int n = sg >> 10, sl = sg & (S_ - 1);
    size_t idx = ((size_t)n * C_ + o) * S_ + sl;
    out[idx] = x[idx] + acc[r] + fb;
  }
}

// ---------------------------------------------------------------------------
extern "C" void kernel_launch(void* const* d_in, const int* in_sizes, int n_in,
                              void* d_out, int out_size, void* d_ws, size_t ws_size,
                              hipStream_t stream) {
  const float* x     = (const float*)d_in[0];
  const float* gamma = (const float*)d_in[1];
  const float* beta  = (const float*)d_in[2];
  const float* wq    = (const float*)d_in[3];
  const float* bq    = (const float*)d_in[4];
  const float* wk    = (const float*)d_in[5];
  const float* bk    = (const float*)d_in[6];
  const float* wv    = (const float*)d_in[7];
  const float* bv    = (const float*)d_in[8];
  const float* wo    = (const float*)d_in[9];
  const float* bo    = (const float*)d_in[10];
  float* out = (float*)d_out;

  char* p = (char*)d_ws;
  const size_t big = (size_t)NS_ * C_ * sizeof(bf16);   // 16 MB each
  const size_t wsz = (size_t)C_ * C_ * sizeof(bf16);    // 512 KB each
  bf16* hB   = (bf16*)p;  p += big;
  bf16* qB   = (bf16*)p;  p += big;
  bf16* kB   = (bf16*)p;  p += big;
  bf16* vTB  = (bf16*)p;  p += big;   // [N][C][S]
  bf16* aoB  = (bf16*)p;  p += big;
  bf16* wqB  = (bf16*)p;  p += wsz;
  bf16* wkB  = (bf16*)p;  p += wsz;
  bf16* wvB  = (bf16*)p;  p += wsz;
  bf16* woB  = (bf16*)p;  p += wsz;
  (void)ws_size; (void)in_sizes; (void)n_in; (void)out_size;

  cvt_w_k<<<dim3((C_ * C_) / 256), dim3(256), 0, stream>>>(
      wq, wk, wv, wo, wqB, wkB, wvB, woB);

  gn_k<<<dim3(N_, G_), dim3(256), 0, stream>>>(x, gamma, beta, hB);

  qkv_k<<<dim3(NS_ / 64, C_ / 16), dim3(128), 0, stream>>>(
      hB, wqB, wkB, wvB, bq, bk, bv, qB, kB, vTB);

  attn_k<<<dim3(N_, S_ / 16), dim3(128), 0, stream>>>(qB, kB, vTB, aoB);

  proj_k<<<dim3(NS_ / 64, C_ / 16), dim3(128), 0, stream>>>(
      aoB, woB, bo, x, out);
}
